// BinaryTreeLSTM_83760452206859
// MI455X (gfx1250) — matrix operand order repaired and checked by reading, not compiled, not run on hardware
//
#include <hip/hip_runtime.h>

// ---------------------------------------------------------------------------
// Binary Tree-LSTM, B=128, 1023 nodes (512 leaves), IN=MEM=300.
// Matrix work done with v_wmma_f32_16x16x32_bf16 (gfx1250 / CDNA5, wave32).
// ---------------------------------------------------------------------------

typedef __attribute__((ext_vector_type(16))) __bf16 v16bf;
typedef __attribute__((ext_vector_type(8)))  float  v8f;

#define BATCH    128
#define NNODES   1023
#define NLEAVES  512
#define INDIM    300
#define MEMDIM   300
#define KPAD_IN  320          // 300 -> pad to multiple of 32
#define KPAD_HH  640          // two children, each padded to 320
#define NPAD_FX  1216         // 1200 -> 76 tiles of 16
#define NPAD_HH  1536         // 1500 -> 96 tiles of 16
#define GSTRIDE  1536         // shared row stride for G scratch
#define CHUNK    64           // nodes per GEMM pass

// workspace layout (bytes)
#define OFF_WFT  ((size_t)0)                                           // [1216][320] bf16
#define OFF_WHT  (OFF_WFT + (size_t)NPAD_FX * KPAD_IN * 2)             // [1536][640] bf16
#define OFF_XBF  (OFF_WHT + (size_t)NPAD_HH * KPAD_HH * 2)             // [512][128][320] bf16
#define OFF_HBF  (OFF_XBF + (size_t)NLEAVES * BATCH * KPAD_IN * 2)     // [1023][128][320] bf16
#define OFF_C    (OFF_HBF + (size_t)NNODES * BATCH * KPAD_IN * 2)      // [1023][128][300] f32
#define OFF_G    (OFF_C   + (size_t)NNODES * BATCH * MEMDIM * 4)       // [64][128][1536] f32

static __device__ __forceinline__ unsigned short f2bf(float f) {
    union { float f; unsigned int u; } v; v.f = f;
    unsigned int r = (v.u + 0x7fffu + ((v.u >> 16) & 1u)) >> 16;  // RNE
    return (unsigned short)r;
}
static __device__ __forceinline__ float sigmoidf(float x) {
    return 1.0f / (1.0f + __expf(-x));
}

// ---------------------------------------------------------------------------
// Weight prep: bf16, transposed (N-major), zero padded.
// WfT[n][k] = Wfioux[k][n]            (n<1200, k<300)
// WhT[n][k] = [Wiouh | Wfh]^T, K rows 0..299 -> k 0..299, 300..599 -> k 320..619
// ---------------------------------------------------------------------------
__global__ void prep_weights(const float* __restrict__ Wfioux,
                             const float* __restrict__ Wiouh,
                             const float* __restrict__ Wfh,
                             unsigned short* __restrict__ WfT,
                             unsigned short* __restrict__ WhT) {
    int idx = blockIdx.x * blockDim.x + threadIdx.x;
    const int totF = NPAD_FX * KPAD_IN;
    const int totH = NPAD_HH * KPAD_HH;
    if (idx < totF) {
        int n = idx / KPAD_IN, k = idx % KPAD_IN;
        float v = (n < 1200 && k < 300) ? Wfioux[(size_t)k * 1200 + n] : 0.0f;
        WfT[idx] = f2bf(v);
    } else if (idx < totF + totH) {
        int id2 = idx - totF;
        int n = id2 / KPAD_HH, k = id2 % KPAD_HH;
        int r = -1;
        if (k < 300) r = k;
        else if (k >= 320 && k < 620) r = k - 20;   // rows 300..599
        float v = 0.0f;
        if (r >= 0 && n < 1500)
            v = (n < 900) ? Wiouh[(size_t)r * 900 + n]
                          : Wfh[(size_t)r * 600 + (n - 900)];
        WhT[id2] = f2bf(v);
    }
}

// ---------------------------------------------------------------------------
// Leaf-input prep: one-shot f32 -> bf16 conversion into padded, GEMM-friendly
// layout Xbf[node][b][320]. 8 elements per thread, no branchy guards.
// ---------------------------------------------------------------------------
__global__ void prep_x(const float* __restrict__ inputs,
                       unsigned short* __restrict__ Xbf) {
    int idx = blockIdx.x * blockDim.x + threadIdx.x;  // one per 8 elements
    const int SEGS = KPAD_IN / 8;                     // 40
    const int total = NLEAVES * BATCH * SEGS;
    if (idx >= total) return;
    int seg  = idx % SEGS;
    int b    = (idx / SEGS) & (BATCH - 1);
    int node = idx / (SEGS * BATCH);
    int d0   = seg * 8;
    const float* src = inputs + ((size_t)b * NNODES + node) * INDIM;

    float f[8];
    if (d0 + 8 <= INDIM) {                            // fast aligned path
        const float4* s4 = (const float4*)(src + d0);
        float4 a = s4[0], c = s4[1];
        f[0] = a.x; f[1] = a.y; f[2] = a.z; f[3] = a.w;
        f[4] = c.x; f[5] = c.y; f[6] = c.z; f[7] = c.w;
    } else {                                          // tail: load then select
#pragma unroll
        for (int e = 0; e < 8; ++e) {
            int d = d0 + e;
            float v = src[d];                         // in-bounds of inputs buffer
            f[e] = (d < INDIM) ? v : 0.0f;
        }
    }
    unsigned short out[8];
#pragma unroll
    for (int e = 0; e < 8; ++e) out[e] = f2bf(f[e]);
    uint4* dst = (uint4*)(Xbf + ((size_t)node * BATCH + b) * KPAD_IN + d0);
    *dst = *(const uint4*)out;
}

// ---------------------------------------------------------------------------
// WMMA fragment loaders (bf16, wave32): A and B both use contiguous k = 16*half + s.
// ---------------------------------------------------------------------------
static __device__ __forceinline__ v16bf load_a_frag(const unsigned short* As, int lane) {
    int m = lane & 15, half = lane >> 4;
    const unsigned int* row = (const unsigned int*)(As + m * 32);
    union { v16bf v; unsigned int u[8]; } r;
#pragma unroll
    for (int j = 0; j < 8; ++j) {
        int k0 = half * 16 + j * 2;
        r.u[j] = row[k0 >> 1];
    }
    return r.v;
}
static __device__ __forceinline__ v16bf load_b_frag(const unsigned short* Bs, int lane) {
    int n = lane & 15, half = lane >> 4;
    const unsigned int* p = (const unsigned int*)(Bs + n * 32 + half * 16);
    union { v16bf v; unsigned int u[8]; } r;
#pragma unroll
    for (int j = 0; j < 8; ++j) r.u[j] = p[j];
    return r.v;
}

// ---------------------------------------------------------------------------
// Tiled GEMM: one block = one node x one 64-wide N strip, M = 128 (batch).
// 8 waves: wave w owns M tile w and all 4 N tiles. A,B staged via LDS.
// LEAF:  A = Xbf[node]            bf16, K = 320
// !LEAF: A = [Hbf[li] | Hbf[ri]]  bf16, K = 640
// ---------------------------------------------------------------------------
template <bool LEAF>
__global__ __launch_bounds__(256)
void gemm_tile_kernel(const unsigned short* __restrict__ Abase,
                      const unsigned short* __restrict__ WT, int kpad,
                      const int* __restrict__ left, const int* __restrict__ right,
                      float* __restrict__ G, int nodeBase) {
    const int cn   = blockIdx.y;
    const int node = nodeBase + cn;
    const int n0   = blockIdx.x * 64;
    const int tid  = threadIdx.x;
    const int lane = tid & 31, wave = tid >> 5;

    int li = 0, ri = 0;
    if (!LEAF) { li = left[node]; ri = right[node]; }

    __shared__ unsigned short As[128 * 32];
    __shared__ unsigned short Bs[64 * 32];

    v8f acc[4];
#pragma unroll
    for (int t = 0; t < 4; ++t)
#pragma unroll
        for (int e = 0; e < 8; ++e) acc[t][e] = 0.0f;

    const int kIters = kpad / 32;
    for (int k0i = 0; k0i < kIters; ++k0i) {
        const int k0 = k0i * 32;

        // ---- stage A tile [128 rows(batch) x 32 k], 32B per thread ----
        {
            const int row = tid >> 1;
            const int seg = (tid & 1) * 16;
            const unsigned short* src;
            if (LEAF) {
                src = Abase + ((size_t)node * BATCH + row) * KPAD_IN + k0 + seg;
            } else {
                const int child = (k0 < KPAD_IN) ? li : ri;
                const int kl = k0 - ((k0 < KPAD_IN) ? 0 : KPAD_IN) + seg;
                src = Abase + ((size_t)child * BATCH + row) * KPAD_IN + kl;
            }
            uint4* d4 = (uint4*)&As[row * 32 + seg];
            const uint4* s4 = (const uint4*)src;
            d4[0] = s4[0]; d4[1] = s4[1];
        }
        // ---- stage B tile [64 n x 32 k], N-major, 16B per thread ----
        {
            const int n   = tid >> 2;
            const int seg = (tid & 3) * 8;
            const unsigned short* src = WT + (size_t)(n0 + n) * kpad + k0 + seg;
            *(uint4*)&Bs[n * 32 + seg] = *(const uint4*)src;
            if (k0i + 1 < kIters) __builtin_prefetch(src + 32, 0, 1);
        }
        __syncthreads();

        v16bf a = load_a_frag(&As[wave * 16 * 32], lane);
#pragma unroll
        for (int t = 0; t < 4; ++t) {
            v16bf b = load_b_frag(&Bs[t * 16 * 32], lane);
            acc[t] = __builtin_amdgcn_wmma_f32_16x16x32_bf16(
                false, a, false, b, (short)0, acc[t], false, false);
        }
        __syncthreads();
    }

    // ---- write out per D-matrix layout: vgpr r -> M = r + half*8 ----
    const int half = lane >> 4, nn = lane & 15;
#pragma unroll
    for (int t = 0; t < 4; ++t) {
        const int col = n0 + t * 16 + nn;
#pragma unroll
        for (int r = 0; r < 8; ++r) {
            const int b = wave * 16 + half * 8 + r;
            G[((size_t)cn * BATCH + b) * GSTRIDE + col] = acc[t][r];
        }
    }
}

// ---------------------------------------------------------------------------
// Leaf epilogue: c = sig(ix)*tanh(ux); h = sig(ox)*tanh(c). G holds x@Wfioux.
// Columns: fx 0..299 | ix 300..599 | ox 600..899 | ux 900..1199.
// ---------------------------------------------------------------------------
__global__ void leaf_epilogue(const float* __restrict__ G, const float* __restrict__ bias,
                              float* __restrict__ C, unsigned short* __restrict__ Hbf,
                              int nodeBase, int count) {
    int idx = blockIdx.x * blockDim.x + threadIdx.x;
    int total = count * BATCH * KPAD_IN;
    if (idx >= total) return;
    int j  = idx % KPAD_IN;
    int b  = (idx / KPAD_IN) & (BATCH - 1);
    int cn = idx / (KPAD_IN * BATCH);
    int node = nodeBase + cn;
    size_t hoff = ((size_t)node * BATCH + b) * KPAD_IN;
    if (j < MEMDIM) {
        const float* g = G + ((size_t)cn * BATCH + b) * GSTRIDE;
        float ig = sigmoidf(bias[300 + j] + g[300 + j]);
        float og = sigmoidf(bias[600 + j] + g[600 + j]);
        float ug = tanhf(bias[900 + j] + g[900 + j]);
        float c  = ig * ug;
        float h  = og * tanhf(c);
        C[((size_t)node * BATCH + b) * MEMDIM + j] = c;
        Hbf[hoff + j] = f2bf(h);
    } else {
        Hbf[hoff + j] = 0;   // zero the K padding
    }
}

// ---------------------------------------------------------------------------
// Internal epilogue. G = ch_h @ [Wiouh|Wfh]:
// ih 0..299 | oh 300..599 | uh 600..899 | fh_left 900..1199 | fh_right 1200..1499
// ---------------------------------------------------------------------------
__global__ void level_epilogue(const float* __restrict__ G, const float* __restrict__ bias,
                               float* __restrict__ C, unsigned short* __restrict__ Hbf,
                               const int* __restrict__ left, const int* __restrict__ right,
                               int nodeBase, int count, float* __restrict__ hout) {
    int idx = blockIdx.x * blockDim.x + threadIdx.x;
    int total = count * BATCH * KPAD_IN;
    if (idx >= total) return;
    int j  = idx % KPAD_IN;
    int b  = (idx / KPAD_IN) & (BATCH - 1);
    int cn = idx / (KPAD_IN * BATCH);
    int node = nodeBase + cn;
    size_t hoff = ((size_t)node * BATCH + b) * KPAD_IN;
    if (j < MEMDIM) {
        int li = left[node], ri = right[node];
        const float* g = G + ((size_t)cn * BATCH + b) * GSTRIDE;
        float ig = sigmoidf(bias[300 + j] + g[j]);
        float og = sigmoidf(bias[600 + j] + g[300 + j]);
        float ug = tanhf(bias[900 + j] + g[600 + j]);
        float fl = sigmoidf(bias[j] + g[900 + j]);
        float fr = sigmoidf(bias[j] + g[1200 + j]);
        float cl = C[((size_t)li * BATCH + b) * MEMDIM + j];
        float cr = C[((size_t)ri * BATCH + b) * MEMDIM + j];
        float c  = ig * ug + fl * cl + fr * cr;
        float h  = og * tanhf(c);
        C[((size_t)node * BATCH + b) * MEMDIM + j] = c;
        Hbf[hoff + j] = f2bf(h);
        if (hout) hout[(size_t)b * MEMDIM + j] = h;   // root h -> output
    } else {
        Hbf[hoff + j] = 0;
    }
}

// ---------------------------------------------------------------------------
extern "C" void kernel_launch(void* const* d_in, const int* in_sizes, int n_in,
                              void* d_out, int out_size, void* d_ws, size_t ws_size,
                              hipStream_t stream) {
    const float* inputs = (const float*)d_in[0];
    const float* Wfioux = (const float*)d_in[1];
    const float* bias   = (const float*)d_in[2];
    const float* Wiouh  = (const float*)d_in[3];
    const float* Wfh    = (const float*)d_in[4];
    const int*   left   = (const int*)d_in[5];
    const int*   right  = (const int*)d_in[6];

    char* ws = (char*)d_ws;
    unsigned short* WfT = (unsigned short*)(ws + OFF_WFT);
    unsigned short* WhT = (unsigned short*)(ws + OFF_WHT);
    unsigned short* Xbf = (unsigned short*)(ws + OFF_XBF);
    unsigned short* Hbf = (unsigned short*)(ws + OFF_HBF);
    float*          C   = (float*)(ws + OFF_C);
    float*          G   = (float*)(ws + OFF_G);

    {   // weight prep
        int tot = NPAD_FX * KPAD_IN + NPAD_HH * KPAD_HH;
        prep_weights<<<(tot + 255) / 256, 256, 0, stream>>>(Wfioux, Wiouh, Wfh, WfT, WhT);
    }
    {   // leaf-input prep (one-shot f32 -> padded bf16)
        int tot = NLEAVES * BATCH * (KPAD_IN / 8);
        prep_x<<<(tot + 255) / 256, 256, 0, stream>>>(inputs, Xbf);
    }

    // ---- leaves (nodes 0..511), chunks of 64 ----
    for (int base = 0; base < NLEAVES; base += CHUNK) {
        dim3 grid(NPAD_FX / 64, CHUNK);
        gemm_tile_kernel<true><<<grid, 256, 0, stream>>>(
            Xbf, WfT, KPAD_IN, nullptr, nullptr, G, base);
        int tot = CHUNK * BATCH * KPAD_IN;
        leaf_epilogue<<<(tot + 255) / 256, 256, 0, stream>>>(G, bias, C, Hbf, base, CHUNK);
    }

    // ---- internal levels (level-order, intra-level parallel) ----
    const int starts[9] = {512, 768, 896, 960, 992, 1008, 1016, 1020, 1022};
    const int counts[9] = {256, 128, 64, 32, 16, 8, 4, 2, 1};
    for (int lv = 0; lv < 9; ++lv) {
        for (int base = 0; base < counts[lv]; base += CHUNK) {
            int cnt = counts[lv] - base; if (cnt > CHUNK) cnt = CHUNK;
            int nodeBase = starts[lv] + base;
            dim3 grid(NPAD_HH / 64, cnt);
            gemm_tile_kernel<false><<<grid, 256, 0, stream>>>(
                Hbf, WhT, KPAD_HH, left, right, G, nodeBase);
            float* hout = (lv == 8) ? ((float*)d_out + (size_t)BATCH * MEMDIM) : nullptr;
            int tot = cnt * BATCH * KPAD_IN;
            level_epilogue<<<(tot + 255) / 256, 256, 0, stream>>>(
                G, bias, C, Hbf, left, right, nodeBase, cnt, hout);
        }
    }

    // root c -> first half of output (C[1022] is [128][300] contiguous)
    hipMemcpyAsync(d_out, ws + OFF_C + (size_t)1022 * BATCH * MEMDIM * 4,
                   (size_t)BATCH * MEMDIM * 4, hipMemcpyDeviceToDevice, stream);
}
